// MyMace_16904991277615
// MI455X (gfx1250) — matrix operand order repaired and checked by reading, not compiled, not run on hardware
//
#include <hip/hip_runtime.h>
#include <cmath>

// ---------------------------------------------------------------------------
// MACE-like equivariant GNN forward on gfx1250 (MI455X).
// Dense compute (edge MLPs + node linears) via v_wmma_f32_16x16x32_f16 with
// f32 accumulation; message passing via fused gather + global f32 atomics
// (accumulators fit in 192MB L2). Memory-bound overall (~2-3GB traffic at
// 23.3 TB/s), so WMMA removes the ~23 GFLOP of dense math from the critical
// path. Edge-feature tiles are staged with GLOBAL_LOAD_ASYNC_TO_LDS_B128
// (one op per wave; 512B tile) when the builtin is available.
// ---------------------------------------------------------------------------

#define NN 20000
#define NE 200000

typedef __attribute__((ext_vector_type(16))) _Float16 v16h;
typedef __attribute__((ext_vector_type(8)))  float    v8f;
typedef __attribute__((ext_vector_type(4)))  int      v4i;

#if defined(__has_builtin)
#  if __has_builtin(__builtin_amdgcn_global_load_async_to_lds_b128)
#    define USE_ASYNC_LDS 1
#  endif
#endif
#ifndef USE_ASYNC_LDS
#  define USE_ASYNC_LDS 0
#endif

#if USE_ASYNC_LDS
// probe-confirmed signature: (int4 AS1*, int4 AS3*, imm offset, imm cpol)
typedef __attribute__((address_space(1))) v4i* as1_v4ip;
typedef __attribute__((address_space(3))) v4i* as3_v4ip;
#endif

__device__ __forceinline__ void wait_async0() {
#if defined(__has_builtin) && __has_builtin(__builtin_amdgcn_s_wait_asynccnt)
  __builtin_amdgcn_s_wait_asynccnt(0);
#else
  asm volatile("s_wait_asynccnt 0x0" ::: "memory");
#endif
}

// constants (constexpr: usable from device code, no attribute needed)
constexpr float kY0   = 0.28209479177387814f; // 1/(2 sqrt(pi))
constexpr float kY1   = 0.48860251190291992f; // sqrt(3/(4 pi))
constexpr float kISQ3 = 0.57735026918962576f;
constexpr float kISQ2 = 0.70710678118654752f;
constexpr float kS8   = 0.35355339059327373f; // 1/sqrt(8)
constexpr float kS64  = 0.125f;               // 1/sqrt(64)
constexpr float kS128 = 0.08838834764831845f; // 1/sqrt(128)
constexpr float kS192 = 0.07216878364870323f; // 1/sqrt(192)
constexpr float kS32  = 0.17677669529663687f; // 1/sqrt(32)

// silu_n via v_rcp_f32 (1 ulp) instead of a correctly-rounded divide:
// saves the v_div_scale/v_div_fmas Newton sequence per activation.
__device__ __forceinline__ float silu_n_dev(float x, float sc) {
  return sc * x * __builtin_amdgcn_rcpf(1.f + __expf(-x));
}

// ---------------------------------------------------------------------------
// Pack a row-major weight W[Cin][Cout] into WMMA-B fragments (f16).
// Tile (kt,nt): per-lane 16 contiguous halfs; B(k,n): lane = n + 16*((k>>3)&1),
// half j = (k&7) + 8*(k>=16) within the 32-row k-chunk.  Rows >= Cin are zero padding.
// ---------------------------------------------------------------------------
__global__ void pack_b_kernel(const float* __restrict__ W, _Float16* __restrict__ out,
                              int Cin, int CinPad, int Cout) {
  int t = blockIdx.x * blockDim.x + threadIdx.x;
  int KT = CinPad >> 5, NT = Cout >> 4;
  int total = KT * NT * 32;
  if (t >= total) return;
  int lane = t & 31, tile = t >> 5;
  int nt = tile % NT, kt = tile / NT;
  int g = lane >> 4, n = lane & 15;
  _Float16* dst = out + ((size_t)tile * 32 + lane) * 16;
  for (int j = 0; j < 16; ++j) {
    int k = kt * 32 + (j & 7) + (g << 3) + ((j >> 3) << 4);
    float w = (k < Cin) ? W[(size_t)k * Cout + nt * 16 + n] : 0.f;
    dst[j] = (_Float16)w;
  }
}

// ---------------------------------------------------------------------------
// A-fragment (16x32 f16, MxK) from an LDS tile stored row-major f32 [16][ld].
// ISA layout: lanes 0-15 row M=lane, K = 0..7,16..23; lanes 16-31 same rows,
// K = 8..15,24..31; VGPR pair packing K=(r&3)*2+c + (r>>2)*16 + g*8.
// ---------------------------------------------------------------------------
__device__ __forceinline__ v16h load_a_frag(const float* tile, int ld, int lane, int k0) {
  v16h a;
  int g = lane >> 4, m = lane & 15;
#pragma unroll
  for (int j = 0; j < 16; ++j) {
    int r = j >> 1, c = j & 1;
    int k = k0 + ((r & 3) << 1) + c + ((r >> 2) << 4) + (g << 3);
    a[j] = (_Float16)tile[m * ld + k];
  }
  return a;
}

// A-fragment for the raw 16x8 feature tile (K padded to 32 with zeros in regs)
__device__ __forceinline__ v16h load_a_frag_feats(const float* fbuf, int lane) {
  v16h a;
  int g = lane >> 4, m = lane & 15;
#pragma unroll
  for (int j = 0; j < 16; ++j) {
    int r = j >> 1, c = j & 1;
    int k = ((r & 3) << 1) + c + ((r >> 2) << 4) + (g << 3);
    a[j] = (k < 8) ? (_Float16)fbuf[m * 8 + k] : (_Float16)0.f;
  }
  return a;
}

// One MLP layer on a 16-row tile, A fragments already in registers.
template <int KT, int NT>
__device__ __forceinline__ void mlp_layer_frags(const v16h* afr, float* out,
                                                const _Float16* __restrict__ Bp, int lane,
                                                float scale, bool dosilu, float siluc) {
  int g = lane >> 4, n = lane & 15;
#pragma unroll
  for (int nt = 0; nt < NT; ++nt) {
    v8f acc = {};
#pragma unroll
    for (int kt = 0; kt < KT; ++kt) {
      v16h b = *((const v16h*)Bp + (size_t)(kt * NT + nt) * 32 + lane);
      acc = __builtin_amdgcn_wmma_f32_16x16x32_f16(false, afr[kt], false, b,
                                                   (short)0, acc, false, false);
    }
#pragma unroll
    for (int r = 0; r < 8; ++r) {
      float v = acc[r] * scale;
      if (dosilu) v = silu_n_dev(v, siluc);
      out[(r + (g << 3)) * (NT * 16) + nt * 16 + n] = v;
    }
  }
}

template <int KT, int NT>
__device__ __forceinline__ void mlp_layer(const float* in, int ldin, float* out,
                                          const _Float16* __restrict__ Bp, int lane,
                                          float scale, bool dosilu, float siluc) {
  v16h afr[KT];
#pragma unroll
  for (int kt = 0; kt < KT; ++kt) afr[kt] = load_a_frag(in, ldin, lane, kt * 32);
  mlp_layer_frags<KT, NT>(afr, out, Bp, lane, scale, dosilu, siluc);
}

// ---------------------------------------------------------------------------
// Edge geometry: dist, permuted Y1 sph-harm, Bessel-poly features (replicates
// the reference exactly, including poly(r=bes/RC) and the (bes<RC) mask).
// ---------------------------------------------------------------------------
__global__ void edge_geom_kernel(const float* __restrict__ vect,
                                 float* __restrict__ y1, float* __restrict__ feats) {
  int e = blockIdx.x * blockDim.x + threadIdx.x;
  if (e >= NE) return;
  float vx = vect[(size_t)e * 3 + 0], vy = vect[(size_t)e * 3 + 1], vz = vect[(size_t)e * 3 + 2];
  float d = sqrtf(vx * vx + vy * vy + vz * vz);
  float inv = 1.f / d;
  y1[(size_t)e * 3 + 0] = kY1 * vy * inv;
  y1[(size_t)e * 3 + 1] = kY1 * vz * inv;
  y1[(size_t)e * 3 + 2] = kY1 * vx * inv;
  const float sq23 = 0.81649658092772603f;      // sqrt(2/3)
  const float pi3  = 1.04719755119659775f;      // pi/3
#pragma unroll
  for (int k = 0; k < 8; ++k) {
    float bes = sq23 * sinf(d * pi3 * (float)(k + 1)) * inv;
    float r = bes * (1.f / 3.f);
    float r2 = r * r, r3 = r2 * r;
    float r6 = r3 * r3, r7 = r6 * r, r8 = r7 * r;
    float poly = 1.f - 28.f * r6 + 48.f * r7 - 21.f * r8;
    feats[(size_t)e * 8 + k] = (bes < 3.f) ? poly : 0.f;
  }
}

// embW[16][64] = emb(10x64) @ W_self(64x64), rows 10..15 zero (tiny, VALU)
__global__ void embw_kernel(const float* __restrict__ emb, const float* __restrict__ Wself,
                            float* __restrict__ embW) {
  int idx = blockIdx.x * blockDim.x + threadIdx.x;   // 1024
  int r = idx >> 6, c = idx & 63;
  float acc = 0.f;
  if (r < 10)
    for (int i = 0; i < 64; ++i) acc += emb[r * 64 + i] * Wself[i * 64 + c];
  embW[idx] = acc;
}

__global__ void xs_kernel(const float* __restrict__ embW, const int* __restrict__ z,
                          float* __restrict__ xs) {
  int idx = blockIdx.x * blockDim.x + threadIdx.x;   // NN*64
  int n = idx >> 6, c = idx & 63;
  xs[idx] = embW[z[n] * 64 + c];
}

// stage this wave's 16x8 f32 feature tile (512B) into LDS
__device__ __forceinline__ void stage_feats(const float* __restrict__ feats, int e0,
                                            float* fbuf, int lane) {
#if USE_ASYNC_LDS
  float* gf = const_cast<float*>(feats) + (size_t)e0 * 8;
  v4i* gg = (v4i*)gf + lane;          // 16B per lane, 512B per wave
  v4i* ll = (v4i*)fbuf + lane;
  __builtin_amdgcn_global_load_async_to_lds_b128((as1_v4ip)gg, (as3_v4ip)ll, 0, 0);
  wait_async0();
#else
  for (int idx = lane; idx < 128; idx += 32)
    fbuf[idx] = feats[(size_t)e0 * 8 + idx];
#endif
}

// ---------------------------------------------------------------------------
// Fused edge pass 1: mlp0 (8->64->64->64->128) + message scatter.
// One wave per 16-edge tile; per-wave LDS ping-pong buffers.
// ---------------------------------------------------------------------------
__global__ void __launch_bounds__(128)
edge_pass1_kernel(const float* __restrict__ feats, const float* __restrict__ y1,
                  const float* __restrict__ xs,
                  const int* __restrict__ ie, const int* __restrict__ je,
                  const _Float16* __restrict__ B1, const _Float16* __restrict__ B2,
                  const _Float16* __restrict__ B3, const _Float16* __restrict__ B4,
                  float* __restrict__ M0, float* __restrict__ M1, float siluc) {
  const int wave = threadIdx.x >> 5, lane = threadIdx.x & 31;
  const int tile = blockIdx.x * 4 + wave;
  const int e0 = tile * 16;
  __shared__ float lds[4][16 * 64 + 16 * 128];
  float* buf0 = &lds[wave][0];          // 16x64
  float* buf1 = &lds[wave][16 * 64];    // 16x128 (also 16x8 feats staging)

  if (e0 + 256 < NE) __builtin_prefetch(&feats[(size_t)(e0 + 256) * 8], 0, 1);

  stage_feats(feats, e0, buf1, lane);
  // mlp0: buf1 -> buf0 -> buf1 -> buf0 -> buf1
  v16h a0 = load_a_frag_feats(buf1, lane);
  mlp_layer_frags<1, 4>(&a0, buf0, B1, lane, kS8, true, siluc);
  mlp_layer<2, 4>(buf0, 64, buf1, B2, lane, kS64, true,  siluc);
  mlp_layer<2, 4>(buf1, 64, buf0, B3, lane, kS64, true,  siluc);
  mlp_layer<2, 8>(buf0, 64, buf1, B4, lane, kS64, false, siluc);  // 16x128: w0 | w1

  // messages: lane handles channels c = lane and c = lane+32 for each edge
  for (int e = 0; e < 16; ++e) {
    int eg = e0 + e;
    int i = ie[eg], j = je[eg];                               // wave-uniform
    float y10 = y1[(size_t)eg * 3 + 0], y11 = y1[(size_t)eg * 3 + 1], y12 = y1[(size_t)eg * 3 + 2];
#pragma unroll
    for (int hh = 0; hh < 2; ++hh) {
      int c = (hh << 5) + lane;
      float xe = xs[(size_t)i * 64 + c];
      float w0 = buf1[e * 128 + c];
      float w1 = buf1[e * 128 + 64 + c];
      atomicAdd(&M0[(size_t)j * 64 + c], xe * kY0 * w0);
      float a = xe * w1;
      atomicAdd(&M1[(size_t)j * 192 + c * 3 + 0], a * y10);
      atomicAdd(&M1[(size_t)j * 192 + c * 3 + 1], a * y11);
      atomicAdd(&M1[(size_t)j * 192 + c * 3 + 2], a * y12);
    }
  }
}

// ---------------------------------------------------------------------------
// Fused edge pass 2: mlp1 (8->64->64->64->320) + equivariant message scatter.
// ---------------------------------------------------------------------------
__global__ void __launch_bounds__(64)
edge_pass2_kernel(const float* __restrict__ feats, const float* __restrict__ y1,
                  const float* __restrict__ sL, const float* __restrict__ vL,
                  const int* __restrict__ ie, const int* __restrict__ je,
                  const _Float16* __restrict__ B1, const _Float16* __restrict__ B2,
                  const _Float16* __restrict__ B3, const _Float16* __restrict__ B4,
                  float* __restrict__ M0b, float* __restrict__ M1b, float siluc) {
  const int wave = threadIdx.x >> 5, lane = threadIdx.x & 31;
  const int tile = blockIdx.x * 2 + wave;
  const int e0 = tile * 16;
  __shared__ float lds[2][16 * 64 + 16 * 320];
  float* buf0 = &lds[wave][0];          // 16x64
  float* buf1 = &lds[wave][16 * 64];    // 16x320

  if (e0 + 128 < NE) __builtin_prefetch(&feats[(size_t)(e0 + 128) * 8], 0, 1);

  stage_feats(feats, e0, buf1, lane);
  v16h a0 = load_a_frag_feats(buf1, lane);
  mlp_layer_frags<1, 4>(&a0, buf0, B1, lane, kS8, true, siluc);
  mlp_layer<2, 4>(buf0, 64, buf1, B2, lane, kS64, true,  siluc);
  mlp_layer<2, 4>(buf1, 64, buf0, B3, lane, kS64, true,  siluc);
  mlp_layer<2, 20>(buf0, 64, buf1, B4, lane, kS64, false, siluc); // 16x320 = 5 x C

  for (int e = 0; e < 16; ++e) {
    int eg = e0 + e;
    int i = ie[eg], j = je[eg];                               // wave-uniform
    float y10 = y1[(size_t)eg * 3 + 0], y11 = y1[(size_t)eg * 3 + 1], y12 = y1[(size_t)eg * 3 + 2];
#pragma unroll
    for (int hh = 0; hh < 2; ++hh) {
      int c = (hh << 5) + lane;
      float se = sL[(size_t)i * 64 + c];
      float v0 = vL[(size_t)i * 192 + c * 3 + 0];
      float v1 = vL[(size_t)i * 192 + c * 3 + 1];
      float v2 = vL[(size_t)i * 192 + c * 3 + 2];
      float t0 = buf1[e * 320 + 0 * 64 + c];
      float t1 = buf1[e * 320 + 1 * 64 + c];
      float t2 = buf1[e * 320 + 2 * 64 + c];
      float t3 = buf1[e * 320 + 3 * 64 + c];
      float t4 = buf1[e * 320 + 4 * 64 + c];
      float dvy = v0 * y10 + v1 * y11 + v2 * y12;
      float cr0 = v1 * y12 - v2 * y11;
      float cr1 = v2 * y10 - v0 * y12;
      float cr2 = v0 * y11 - v1 * y10;
      atomicAdd(&M0b[(size_t)j * 128 + c],      se * kY0 * t0);
      atomicAdd(&M0b[(size_t)j * 128 + 64 + c], dvy * kISQ3 * t1);
      float g1 = se * t2, g2 = kY0 * t3, g3 = kISQ2 * t4;
      size_t base = (size_t)j * 576;
      atomicAdd(&M1b[base + c * 3 + 0], g1 * y10);
      atomicAdd(&M1b[base + c * 3 + 1], g1 * y11);
      atomicAdd(&M1b[base + c * 3 + 2], g1 * y12);
      atomicAdd(&M1b[base + (64 + c) * 3 + 0], v0 * g2);
      atomicAdd(&M1b[base + (64 + c) * 3 + 1], v1 * g2);
      atomicAdd(&M1b[base + (64 + c) * 3 + 2], v2 * g2);
      atomicAdd(&M1b[base + (128 + c) * 3 + 0], cr0 * g3);
      atomicAdd(&M1b[base + (128 + c) * 3 + 1], cr1 * g3);
      atomicAdd(&M1b[base + (128 + c) * 3 + 2], cr2 * g3);
    }
  }
}

// ---------------------------------------------------------------------------
// Generic node linear via WMMA: Y[row, n] = act(sum_k X[row, k] * W[k, n] * scale)
// with arbitrary row/col strides (serves lin and each component of lin_v).
// One block (128 thr / 4 waves) per 16-row tile; waves split the Cout tiles.
// ---------------------------------------------------------------------------
__global__ void __launch_bounds__(128)
lin_wmma_kernel(const float* __restrict__ X, int ldx, int xcs,
                const _Float16* __restrict__ Bp,
                float* __restrict__ Y, int ldy, int ycs,
                int cin, int cout, float scale, int act, float siluc) {
  const int wave = threadIdx.x >> 5, lane = threadIdx.x & 31;
  const int row0 = blockIdx.x * 16;
  const int NT = cout >> 4, KT = cin >> 5;
  extern __shared__ float tile[];                     // 16 x cin
  for (int idx = threadIdx.x; idx < 16 * cin; idx += 128) {
    int r = idx / cin, k = idx - r * cin;
    tile[idx] = X[(size_t)(row0 + r) * ldx + (size_t)k * xcs];
  }
  __syncthreads();
  int g = lane >> 4, n = lane & 15;
  for (int nt = wave; nt < NT; nt += 4) {
    v8f acc = {};
    for (int kt = 0; kt < KT; ++kt) {
      v16h a = load_a_frag(tile, cin, lane, kt * 32);
      v16h b = *((const v16h*)Bp + (size_t)(kt * NT + nt) * 32 + lane);
      acc = __builtin_amdgcn_wmma_f32_16x16x32_f16(false, a, false, b,
                                                   (short)0, acc, false, false);
    }
#pragma unroll
    for (int r = 0; r < 8; ++r) {
      float v = acc[r] * scale;
      if (act) v = silu_n_dev(v, siluc);
      Y[(size_t)(row0 + r + (g << 3)) * ldy + (size_t)(nt * 16 + n) * ycs] = v;
    }
  }
}

// sym_contract: per (node, channel) elementwise with per-species coefficients
__global__ void sym_contract_kernel(const float* __restrict__ s, const float* __restrict__ v,
                                    const float* __restrict__ wsp, const float* __restrict__ wvp,
                                    const int* __restrict__ z,
                                    float* __restrict__ so, float* __restrict__ vo) {
  int idx = blockIdx.x * blockDim.x + threadIdx.x;   // NN*64
  int n = idx >> 6, c = idx & 63;
  int zz = z[n];
  const float* a = wsp + (size_t)zz * 320;
  const float* b = wvp + (size_t)zz * 256;
  float sv = s[idx];
  float v0 = v[(size_t)n * 192 + c * 3 + 0];
  float v1 = v[(size_t)n * 192 + c * 3 + 1];
  float v2 = v[(size_t)n * 192 + c * 3 + 2];
  float vv = (v0 * v0 + v1 * v1 + v2 * v2) * kISQ3;
  float s2 = sv * sv;
  so[idx] = a[c] * sv + a[64 + c] * s2 + a[128 + c] * vv + a[192 + c] * s2 * sv + a[256 + c] * sv * vv;
  float gate = b[c] + b[64 + c] * sv + b[128 + c] * s2 + b[192 + c] * vv;
  vo[(size_t)n * 192 + c * 3 + 0] = gate * v0;
  vo[(size_t)n * 192 + c * 3 + 1] = gate * v1;
  vo[(size_t)n * 192 + c * 3 + 2] = gate * v2;
}

// equiv = concat([s[:,:,None], v], -1)  ->  d_out[1 + (n*64+c)*4 + {0..3}]
__global__ void equiv_kernel(const float* __restrict__ s, const float* __restrict__ v,
                             float* __restrict__ out) {
  int idx = blockIdx.x * blockDim.x + threadIdx.x;   // NN*64
  int n = idx >> 6, c = idx & 63;
  float* o = out + 1 + (size_t)idx * 4;
  o[0] = s[idx];
  o[1] = v[(size_t)n * 192 + c * 3 + 0];
  o[2] = v[(size_t)n * 192 + c * 3 + 1];
  o[3] = v[(size_t)n * 192 + c * 3 + 2];
}

// out[0] += sum_n  h[n,:] . Wf2 * 1/sqrt(32)   (block-reduced, 1 atomic/block)
__global__ void head_final_kernel(const float* __restrict__ h, const float* __restrict__ Wf2,
                                  float* __restrict__ out) {
  __shared__ float red[256];
  int n = blockIdx.x * 256 + threadIdx.x;
  float acc = 0.f;
  if (n < NN) {
    for (int c = 0; c < 32; ++c) acc += h[(size_t)n * 32 + c] * Wf2[c];
    acc *= kS32;
  }
  red[threadIdx.x] = acc;
  __syncthreads();
  for (int s = 128; s > 0; s >>= 1) {
    if (threadIdx.x < s) red[threadIdx.x] += red[threadIdx.x + s];
    __syncthreads();
  }
  if (threadIdx.x == 0) atomicAdd(out, red[0]);
}

// ---------------------------------------------------------------------------
// Host side
// ---------------------------------------------------------------------------
static float compute_silu_c() {   // same trapezoid rule as the reference
  const int NPT = 100001;
  const double lo = -8.0, hi = 8.0, dx = (hi - lo) / (NPT - 1);
  double sum = 0.0, prev = 0.0;
  for (int i = 0; i < NPT; ++i) {
    double x = lo + i * dx;
    double pdf = exp(-0.5 * x * x) / sqrt(2.0 * M_PI);
    double sl = x / (1.0 + exp(-x));
    double f = sl * sl * pdf;
    if (i) sum += 0.5 * (f + prev) * dx;
    prev = f;
  }
  return (float)(1.0 / sqrt(sum));
}

extern "C" void kernel_launch(void* const* d_in, const int* in_sizes, int n_in,
                              void* d_out, int out_size, void* d_ws, size_t ws_size,
                              hipStream_t stream) {
  // params in setup_inputs() insertion order, then vect / atomic_numbers / edges
  const float* emb    = (const float*)d_in[0];
  const float* Wself  = (const float*)d_in[1];
  const float* mlp0w[4] = {(const float*)d_in[2], (const float*)d_in[3],
                           (const float*)d_in[4], (const float*)d_in[5]};
  const float* li_W0  = (const float*)d_in[6];
  const float* li_W1  = (const float*)d_in[7];
  const float* sc0_ws = (const float*)d_in[8];
  const float* sc0_wv = (const float*)d_in[9];
  const float* lp0_W0 = (const float*)d_in[10];
  const float* lp0_W1 = (const float*)d_in[11];
  const float* lu_W0  = (const float*)d_in[12];
  const float* lu_W1  = (const float*)d_in[13];
  const float* mlp1w[4] = {(const float*)d_in[14], (const float*)d_in[15],
                           (const float*)d_in[16], (const float*)d_in[17]};
  const float* il_W0  = (const float*)d_in[18];
  const float* il_W1  = (const float*)d_in[19];
  const float* sc1_ws = (const float*)d_in[20];
  const float* sc1_wv = (const float*)d_in[21];
  const float* lp1_W0 = (const float*)d_in[22];
  const float* lp1_W1 = (const float*)d_in[23];
  const float* Wf1    = (const float*)d_in[24];
  const float* Wf2    = (const float*)d_in[25];
  const float* vect   = (const float*)d_in[26];
  const int*   zat    = (const int*)d_in[27];
  const int*   ie     = (const int*)d_in[28];
  const int*   je     = (const int*)d_in[29];
  (void)in_sizes; (void)n_in; (void)out_size; (void)ws_size;

  const float SILU_C = compute_silu_c();
  float* outp = (float*)d_out;

  // workspace layout (floats)
  float* wsf = (float*)d_ws;
  size_t off = 0;
  float* y1    = wsf + off; off += (size_t)NE * 3;
  float* feats = wsf + off; off += (size_t)NE * 8;
  float* embW  = wsf + off; off += 1024;
  float* xs    = wsf + off; off += (size_t)NN * 64;
  float* M0a   = wsf + off; off += (size_t)NN * 64;
  float* M1a   = wsf + off; off += (size_t)NN * 192;
  float* M0b   = wsf + off; off += (size_t)NN * 128;
  float* M1b   = wsf + off; off += (size_t)NN * 576;
  float* s0    = wsf + off; off += (size_t)NN * 64;
  float* s1    = wsf + off; off += (size_t)NN * 64;
  float* v0    = wsf + off; off += (size_t)NN * 192;
  float* v1    = wsf + off; off += (size_t)NN * 192;
  float* hb    = wsf + off; off += (size_t)NN * 32;
  _Float16* pbase = (_Float16*)(wsf + off);

  // packed-weight sub-allocations (halfs)
  size_t po = 0;
  auto palloc = [&](int KT, int NT) { _Float16* p = pbase + po; po += (size_t)KT * NT * 512; return p; };
  _Float16* P_m0[4] = {palloc(1, 4), palloc(2, 4), palloc(2, 4), palloc(2, 8)};
  _Float16* P_m1[4] = {palloc(1, 4), palloc(2, 4), palloc(2, 4), palloc(2, 20)};
  _Float16* P_li0 = palloc(2, 4);  _Float16* P_li1 = palloc(2, 4);
  _Float16* P_lp00 = palloc(2, 4); _Float16* P_lp01 = palloc(2, 4);
  _Float16* P_lu0 = palloc(2, 4);  _Float16* P_lu1 = palloc(2, 4);
  _Float16* P_il0 = palloc(4, 4);  _Float16* P_il1 = palloc(6, 4);
  _Float16* P_lp10 = palloc(2, 4); _Float16* P_lp11 = palloc(2, 4);
  _Float16* P_wf1 = palloc(2, 2);

  // zero accumulators + scalar output
  (void)hipMemsetAsync(M0a, 0, (size_t)NN * 64 * 4, stream);
  (void)hipMemsetAsync(M1a, 0, (size_t)NN * 192 * 4, stream);
  (void)hipMemsetAsync(M0b, 0, (size_t)NN * 128 * 4, stream);
  (void)hipMemsetAsync(M1b, 0, (size_t)NN * 576 * 4, stream);
  (void)hipMemsetAsync(outp, 0, 4, stream);

  auto pack = [&](const float* W, int Cin, int CinPad, int Cout, _Float16* dst) {
    int total = (CinPad >> 5) * (Cout >> 4) * 32;
    pack_b_kernel<<<(total + 255) / 256, 256, 0, stream>>>(W, dst, Cin, CinPad, Cout);
  };
  pack(mlp0w[0], 8, 32, 64, P_m0[0]);  pack(mlp0w[1], 64, 64, 64, P_m0[1]);
  pack(mlp0w[2], 64, 64, 64, P_m0[2]); pack(mlp0w[3], 64, 64, 128, P_m0[3]);
  pack(mlp1w[0], 8, 32, 64, P_m1[0]);  pack(mlp1w[1], 64, 64, 64, P_m1[1]);
  pack(mlp1w[2], 64, 64, 64, P_m1[2]); pack(mlp1w[3], 64, 64, 320, P_m1[3]);
  pack(li_W0, 64, 64, 64, P_li0);      pack(li_W1, 64, 64, 64, P_li1);
  pack(lp0_W0, 64, 64, 64, P_lp00);    pack(lp0_W1, 64, 64, 64, P_lp01);
  pack(lu_W0, 64, 64, 64, P_lu0);      pack(lu_W1, 64, 64, 64, P_lu1);
  pack(il_W0, 128, 128, 64, P_il0);    pack(il_W1, 192, 192, 64, P_il1);
  pack(lp1_W0, 64, 64, 64, P_lp10);    pack(lp1_W1, 64, 64, 64, P_lp11);
  pack(Wf1, 64, 64, 32, P_wf1);

  // geometry + embedding table + xs gather
  edge_geom_kernel<<<(NE + 255) / 256, 256, 0, stream>>>(vect, y1, feats);
  embw_kernel<<<4, 256, 0, stream>>>(emb, Wself, embW);
  xs_kernel<<<NN * 64 / 256, 256, 0, stream>>>(embW, zat, xs);

  // pass 1: mlp0 + scatter (12500 edge tiles, 4 waves per block)
  edge_pass1_kernel<<<NE / 16 / 4, 128, 0, stream>>>(
      feats, y1, xs, ie, je, P_m0[0], P_m0[1], P_m0[2], P_m0[3], M0a, M1a, SILU_C);

  auto lin = [&](const float* X, int ldx, int xcs, _Float16* Bp, float* Y, int ldy, int ycs,
                 int cin, int cout, float scale, int act) {
    lin_wmma_kernel<<<NN / 16, 128, 16 * cin * 4, stream>>>(X, ldx, xcs, Bp, Y, ldy, ycs,
                                                            cin, cout, scale, act, SILU_C);
  };
  auto lin_v = [&](const float* X, int ldx, _Float16* Bp, float* Y, int cin, float scale) {
    for (int m = 0; m < 3; ++m)
      lin(X + m, ldx, 3, Bp, Y + m, 192, 3, cin, 64, scale, 0);
  };

  // li
  lin(M0a, 64, 1, P_li0, s0, 64, 1, 64, 64, kS64, 0);
  lin_v(M1a, 192, P_li1, v0, 64, kS64);
  // sym_contract 0
  sym_contract_kernel<<<NN * 64 / 256, 256, 0, stream>>>(s0, v0, sc0_ws, sc0_wv, zat, s1, v1);
  // lp0
  lin(s1, 64, 1, P_lp00, s0, 64, 1, 64, 64, kS64, 0);
  lin_v(v1, 192, P_lp01, v0, 64, kS64);
  // lu
  lin(s0, 64, 1, P_lu0, s1, 64, 1, 64, 64, kS64, 0);
  lin_v(v0, 192, P_lu1, v1, 64, kS64);

  // pass 2: mlp1 + equivariant scatter (12500 tiles, 2 waves per block)
  edge_pass2_kernel<<<NE / 16 / 2, 64, 0, stream>>>(
      feats, y1, s1, v1, ie, je, P_m1[0], P_m1[1], P_m1[2], P_m1[3], M0b, M1b, SILU_C);

  // il
  lin(M0b, 128, 1, P_il0, s0, 64, 1, 128, 64, kS128, 0);
  lin_v(M1b, 576, P_il1, v0, 192, kS192);
  // sym_contract 1
  sym_contract_kernel<<<NN * 64 / 256, 256, 0, stream>>>(s0, v0, sc1_ws, sc1_wv, zat, s1, v1);
  // lp1
  lin(s1, 64, 1, P_lp10, s0, 64, 1, 64, 64, kS64, 0);
  lin_v(v1, 192, P_lp11, v0, 64, kS64);

  // outputs
  equiv_kernel<<<NN * 64 / 256, 256, 0, stream>>>(s0, v0, outp);
  lin(s0, 64, 1, P_wf1, hb, 32, 1, 64, 32, kS64, 1);   // h = silu_n(lin(s, Wf1))
  head_final_kernel<<<(NN + 255) / 256, 256, 0, stream>>>(hb, Wf2, outp);
}
